// QAConv_72559177499038
// MI455X (gfx1250) — matrix operand order repaired and checked by reading, not compiled, not run on hardware
//
#include <hip/hip_runtime.h>

typedef __bf16 bf16;
typedef __attribute__((ext_vector_type(16))) __bf16 v16bf;
typedef __attribute__((ext_vector_type(8)))  __bf16 v8bf;
typedef __attribute__((ext_vector_type(8)))  float  v8f;

#define NP   64
#define NG   64
#define NC   512
#define HW   192
#define KC   32
#define EPSV 1e-5f
#define NEG_BIG (-3.402823466e38f)

struct __align__(128) Smem {
    bf16  A[HW * KC];        // gf chunk, [r][k] (k contiguous)  12 KB
    bf16  B[HW * KC];        // pf chunk, [s][k] (k contiguous)  12 KB
    float colPart[4][HW];    // per wave-row partial max over r  (indexed by s)
    float rowPart[2][HW];    // per wave-col partial max over s  (indexed by r)
    float sb[8];             // block reduction scratch
};

__device__ __forceinline__ float blockSum(float v, float* sb) {
#pragma unroll
    for (int off = 16; off; off >>= 1) v += __shfl_xor(v, off);
    if ((threadIdx.x & 31) == 0) sb[threadIdx.x >> 5] = v;
    __syncthreads();
    float r = 0.f;
    if (threadIdx.x == 0)
        for (int i = 0; i < 8; ++i) r += sb[i];
    __syncthreads();
    return r;   // valid on thread 0 only
}

// Kernel 1: per-(p,g) 192x192x512 GEMM via bf16 WMMA + dual-axis max pool.
__global__ __launch_bounds__(256) void qaconv_gemm_pool(
    const float* __restrict__ pf, const float* __restrict__ gf,
    const float* __restrict__ fcw,
    float* __restrict__ Twt, float* __restrict__ SumM, float* __restrict__ SumM2)
{
    __shared__ Smem sh;
    const int tid  = threadIdx.x;
    const int pg   = blockIdx.x;
    const int p    = pg / NG;
    const int g    = pg % NG;
    const float* __restrict__ pfBase = pf + (size_t)p * NC * HW;
    const float* __restrict__ gfBase = gf + (size_t)g * NC * HW;

    const int lane = tid & 31;
    const int wave = tid >> 5;
    const int wr   = wave >> 1;      // 0..3 : 48-row strip
    const int wc   = wave & 1;       // 0..1 : 96-col strip
    const int half = (lane >> 4) & 1;
    const int l15  = lane & 15;

    // division-free staging walk: idx4 = tid + 256*it over 1536 float4 slots,
    // k = idx4 / 48 (K slice), rv = idx4 % 48 (float4 column); 256 = 5*48 + 16.
    const int k0  = tid / 48;
    const int rv0 = tid - k0 * 48;

    v8f acc[3][6];
#pragma unroll
    for (int rt = 0; rt < 3; ++rt)
#pragma unroll
        for (int st = 0; st < 6; ++st)
            acc[rt][st] = (v8f){0.f,0.f,0.f,0.f,0.f,0.f,0.f,0.f};

    for (int kc = 0; kc < NC; kc += KC) {
        // ---- cooperative stage: f32 global (b128) -> bf16 LDS, transposed to [row][k] ----
        {
            int k = k0, rv = rv0;
#pragma unroll
            for (int it = 0; it < 6; ++it) {
                const float4 ga = *(const float4*)&gfBase[(size_t)(kc + k) * HW + rv * 4];
                const float4 pb = *(const float4*)&pfBase[(size_t)(kc + k) * HW + rv * 4];
                const int r = rv * 4;
                sh.A[(r + 0) * KC + k] = (bf16)ga.x;
                sh.A[(r + 1) * KC + k] = (bf16)ga.y;
                sh.A[(r + 2) * KC + k] = (bf16)ga.z;
                sh.A[(r + 3) * KC + k] = (bf16)ga.w;
                sh.B[(r + 0) * KC + k] = (bf16)pb.x;
                sh.B[(r + 1) * KC + k] = (bf16)pb.y;
                sh.B[(r + 2) * KC + k] = (bf16)pb.z;
                sh.B[(r + 3) * KC + k] = (bf16)pb.w;
                k += 5; rv += 16;
                if (rv >= 48) { rv -= 48; ++k; }
            }
        }
        if (kc + KC < NC) {                // hint next chunk toward L2
            __builtin_prefetch(&gfBase[(size_t)(kc + KC) * HW + tid * 24], 0, 0);
            __builtin_prefetch(&pfBase[(size_t)(kc + KC) * HW + tid * 24], 0, 0);
        }
        __syncthreads();

        // ---- fragment loads (contiguous k order for A and B: lanes 0-15 K{0-15}, 16-31 K{16-31}) ----
        v16bf afrag[3];
#pragma unroll
        for (int rt = 0; rt < 3; ++rt) {
            int r = wr * 48 + rt * 16 + l15;
            const bf16* pa = &sh.A[r * KC + half * 16];
            v8bf lo = *(const v8bf*)pa;
            v8bf hi = *(const v8bf*)(pa + 8);
            afrag[rt] = __builtin_shufflevector(lo, hi,
                0,1,2,3,4,5,6,7,8,9,10,11,12,13,14,15);
        }
        v16bf bfrag[6];
#pragma unroll
        for (int st = 0; st < 6; ++st) {
            int s = wc * 96 + st * 16 + l15;
            bfrag[st] = *(const v16bf*)&sh.B[s * KC + half * 16];
        }
#pragma unroll
        for (int st = 0; st < 6; ++st)
#pragma unroll
            for (int rt = 0; rt < 3; ++rt)
                acc[rt][st] = __builtin_amdgcn_wmma_f32_16x16x32_bf16(
                    false, afrag[rt], false, bfrag[st],
                    (short)0, acc[rt][st], false, false);
        __syncthreads();
    }

    // ---- dual-axis max pooling using the C/D VGPR layout ----
    // column max (over M=r) -> maxOverR indexed by s
#pragma unroll
    for (int st = 0; st < 6; ++st) {
        float cm = NEG_BIG;
#pragma unroll
        for (int rt = 0; rt < 3; ++rt)
#pragma unroll
            for (int j = 0; j < 8; ++j)
                cm = fmaxf(cm, acc[rt][st][j]);
        cm = fmaxf(cm, __shfl_xor(cm, 16));      // combine M-halves
        if (lane < 16)
            sh.colPart[wr][wc * 96 + st * 16 + lane] = cm;
    }
    // row max (over N=s) -> maxOverS indexed by r
#pragma unroll
    for (int rt = 0; rt < 3; ++rt) {
#pragma unroll
        for (int j = 0; j < 8; ++j) {
            float rm = NEG_BIG;
#pragma unroll
            for (int st = 0; st < 6; ++st)
                rm = fmaxf(rm, acc[rt][st][j]);
            rm = fmaxf(rm, __shfl_xor(rm, 1));
            rm = fmaxf(rm, __shfl_xor(rm, 2));
            rm = fmaxf(rm, __shfl_xor(rm, 4));
            rm = fmaxf(rm, __shfl_xor(rm, 8));
            if (l15 == 0)
                sh.rowPart[wc][wr * 48 + rt * 16 + half * 8 + j] = rm;
        }
    }
    __syncthreads();

    float tw = 0.f, sm = 0.f, sq = 0.f;
    if (tid < HW) {
        float mR = fmaxf(fmaxf(sh.colPart[0][tid], sh.colPart[1][tid]),
                         fmaxf(sh.colPart[2][tid], sh.colPart[3][tid]));
        float mS = fmaxf(sh.rowPart[0][tid], sh.rowPart[1][tid]);
        float w  = fcw[tid];
        tw = w * (mR + mS);          // weighted sum (FC)
        sm = mR + mS;                // sum for BN1 mean
        sq = mR * mR + mS * mS;      // sum of squares for BN1 var
    }
    tw = blockSum(tw, sh.sb);
    sm = blockSum(sm, sh.sb);
    sq = blockSum(sq, sh.sb);
    if (tid == 0) {
        Twt[pg]   = tw;
        SumM[pg]  = sm;
        SumM2[pg] = sq;
    }
}

// Kernel 2: reduce per-pair sums -> BN scale/shift scalars.
__global__ __launch_bounds__(256) void qaconv_stats(
    const float* __restrict__ Twt, const float* __restrict__ SumM,
    const float* __restrict__ SumM2,
    const float* __restrict__ bnw, const float* __restrict__ lbnw,
    const float* __restrict__ lbnb,
    float* __restrict__ scalars)
{
    __shared__ float sb[8];
    float tT = 0.f, tT2 = 0.f, tM = 0.f, tM2 = 0.f;
    for (int i = threadIdx.x; i < NP * NG; i += 256) {
        float t = Twt[i];
        tT += t; tT2 += t * t;
        tM += SumM[i]; tM2 += SumM2[i];
    }
    tT  = blockSum(tT,  sb);
    tT2 = blockSum(tT2, sb);
    tM  = blockSum(tM,  sb);
    tM2 = blockSum(tM2, sb);
    if (threadIdx.x == 0) {
        const float Ntot = 2.0f * NP * NG * HW;       // BN1 element count
        float mu1  = tM / Ntot;
        float v1   = tM2 / Ntot - mu1 * mu1;
        float c1   = rsqrtf(v1 + EPSV) * bnw[0];      // BN1 slope (bias cancels in BN2)
        const float Npg = (float)(NP * NG);
        float meanT = tT / Npg;
        float varT  = tT2 / Npg - meanT * meanT;
        float A = c1 * lbnw[0] * rsqrtf(c1 * c1 * varT + EPSV);
        scalars[0] = meanT;
        scalars[1] = A;
        scalars[2] = lbnb[0];
    }
}

// Kernel 3: elementwise finalize.
__global__ void qaconv_finalize(const float* __restrict__ Twt,
                                const float* __restrict__ scalars,
                                float* __restrict__ out, int n)
{
    int i = blockIdx.x * blockDim.x + threadIdx.x;
    if (i < n)
        out[i] = (Twt[i] - scalars[0]) * scalars[1] + scalars[2];
}

extern "C" void kernel_launch(void* const* d_in, const int* in_sizes, int n_in,
                              void* d_out, int out_size, void* d_ws, size_t ws_size,
                              hipStream_t stream)
{
    (void)in_sizes; (void)n_in; (void)out_size; (void)ws_size;
    const float* pf   = (const float*)d_in[0];   // [64,512,24,8]
    const float* gf   = (const float*)d_in[1];   // [64,512,24,8]
    const float* bnw  = (const float*)d_in[2];   // bn_weight[1]
    // d_in[3] bn_bias: cancels through final BN
    const float* fcw  = (const float*)d_in[4];   // fc_weight[1,192]
    // d_in[5] fc_bias: cancels through final BN
    const float* lbnw = (const float*)d_in[6];   // lbn_weight[1]
    const float* lbnb = (const float*)d_in[7];   // lbn_bias[1]

    float* ws      = (float*)d_ws;
    float* Twt     = ws;                 // 4096
    float* SumM    = ws + 4096;          // 4096
    float* SumM2   = ws + 8192;          // 4096
    float* scalars = ws + 12288;         // 3

    qaconv_gemm_pool<<<NP * NG, 256, 0, stream>>>(pf, gf, fcw, Twt, SumM, SumM2);
    qaconv_stats<<<1, 256, 0, stream>>>(Twt, SumM, SumM2, bnw, lbnw, lbnb, scalars);
    qaconv_finalize<<<(NP * NG + 255) / 256, 256, 0, stream>>>(Twt, scalars,
                                                               (float*)d_out, NP * NG);
}
